// MultiSelfAttentionHeadLocal_53609781789411
// MI455X (gfx1250) — matrix operand order repaired and checked by reading, not compiled, not run on hardware
//
#include <hip/hip_runtime.h>
#include <hip/hip_bf16.h>

// MultiSelfAttentionHeadLocal for MI455X (gfx1250, wave32, WMMA + TDM).
// B=4, S=8192, D=1024, H=16, Dh=64, BLOCK=256, nb=32, M=B*S=32768.
// All GEMMs run on v_wmma_f32_16x16x32_bf16 (bf16 in, f32 accumulate).
// Attention K-tiles are staged to LDS with the Tensor Data Mover.

typedef __bf16 bf16;
typedef __attribute__((ext_vector_type(16))) __bf16 v16bf;
typedef __attribute__((ext_vector_type(8)))  float  v8f;

#define WMMA_BF16 __builtin_amdgcn_wmma_f32_16x16x32_bf16

#if defined(__has_builtin)
#if __has_builtin(__builtin_amdgcn_tensor_load_to_lds) && \
    __has_builtin(__builtin_amdgcn_s_wait_tensorcnt)
#define HAVE_TDM 1
#endif
#endif
#ifndef HAVE_TDM
#define HAVE_TDM 0
#endif

#if HAVE_TDM
typedef __attribute__((ext_vector_type(4))) unsigned int u32x4;
typedef __attribute__((ext_vector_type(8))) int          i32x8;
typedef __attribute__((ext_vector_type(4))) int          i32x4;
#endif

union FragU { v16bf v; uint4 q[2]; };

// A-matrix 16x32 (MxK) bf16 fragment, per CDNA5 ISA layout:
// lane<16: M=lane,    elements 0..7 -> K=0..7,  8..15 -> K=16..23
// lane>=16:M=lane-16, elements 0..7 -> K=8..15, 8..15 -> K=24..31
// Source must be row-major with contiguous K, leading dim ldk (elements).
__device__ __forceinline__ v16bf load_fragA(const bf16* base, int ldk, int lane) {
  const int r  = lane & 15;
  const int kb = (lane >> 4) << 3;        // 0 or 8
  const bf16* p = base + r * ldk + kb;
  FragU f;
  f.q[0] = *(const uint4*)(p);            // K = kb .. kb+7
  f.q[1] = *(const uint4*)(p + 16);       // K = kb+16 .. kb+23
  return f.v;
}

// B-matrix 32x16 (KxN) bf16 fragment:
// lane<16: N=lane, elements 0..7 -> K=0..7, 8..15 -> K=16..23; lane>=16: N=lane-16, K=8..15, 24..31.
// Source is N-major rows with contiguous K (transposed operand of an NT GEMM).
__device__ __forceinline__ v16bf load_fragB(const bf16* base, int ldk, int lane) {
  const int n  = lane & 15;
  const int k0 = (lane >> 4) << 3;        // 0 or 8
  const bf16* p = base + n * ldk + k0;
  FragU f;
  f.q[0] = *(const uint4*)(p);
  f.q[1] = *(const uint4*)(p + 16);
  return f.v;
}

// ---------------------------------------------------------------- kernel 1
// LayerNorm over D=1024, cast to bf16. One 256-thread block per row.
__global__ __launch_bounds__(256)
void ln_cast_kernel(const float* __restrict__ x, bf16* __restrict__ hout) {
  const int row = blockIdx.x;
  const float* xr = x + (size_t)row * 1024;
  float v[4];
  float s = 0.f;
#pragma unroll
  for (int i = 0; i < 4; ++i) { v[i] = xr[threadIdx.x + 256 * i]; s += v[i]; }
  __shared__ float red[8];
  for (int off = 16; off >= 1; off >>= 1) s += __shfl_xor(s, off, 32);
  const int wid = threadIdx.x >> 5, lid = threadIdx.x & 31;
  if (lid == 0) red[wid] = s;
  __syncthreads();
  float tot = 0.f;
#pragma unroll
  for (int i = 0; i < 8; ++i) tot += red[i];
  const float mean = tot * (1.0f / 1024.0f);
  float vs = 0.f;
#pragma unroll
  for (int i = 0; i < 4; ++i) { float d = v[i] - mean; vs += d * d; }
  __syncthreads();
  for (int off = 16; off >= 1; off >>= 1) vs += __shfl_xor(vs, off, 32);
  if (lid == 0) red[wid] = vs;
  __syncthreads();
  float vtot = 0.f;
#pragma unroll
  for (int i = 0; i < 8; ++i) vtot += red[i];
  const float rstd = rsqrtf(vtot * (1.0f / 1024.0f) + 1e-5f);
  bf16* hr = hout + (size_t)row * 1024;
#pragma unroll
  for (int i = 0; i < 4; ++i)
    hr[threadIdx.x + 256 * i] = (bf16)((v[i] - mean) * rstd);
}

// ---------------------------------------------------------------- kernel 2
// Convert the four 1024x1024 weights to bf16: out = [Wq|Wk|Wv|Wo] flat.
__global__ __launch_bounds__(256)
void wcast_kernel(const float* __restrict__ wq, const float* __restrict__ wk,
                  const float* __restrict__ wv, const float* __restrict__ wo,
                  bf16* __restrict__ out) {
  const size_t total = (size_t)4 * 1024 * 1024;
  for (size_t idx = (size_t)blockIdx.x * 256 + threadIdx.x; idx < total;
       idx += (size_t)gridDim.x * 256) {
    const int w = (int)(idx >> 20);
    const size_t off = idx & ((1u << 20) - 1);
    const float* sp = (w == 0) ? wq : (w == 1) ? wk : (w == 2) ? wv : wo;
    out[idx] = (bf16)sp[off];
  }
}

// ---------------------------------------------------------------- kernel 3
// QKV projection: C = h(bf16)[M,1024] x W^T (NT GEMM), M=32768.
// blockIdx.z selects {q,k,v}. q,k stored f32 in [B,H,S,64] (pre head-LN);
// v stored bf16 transposed per block: vT[B,H,blk,64,256].
// Workgroup tile 128x128, 8 waves as 4(M)x2(N), wave tile 32x64, K-chunk 32.
__global__ __launch_bounds__(256)
void qkv_gemm_kernel(const bf16* __restrict__ hbf, const bf16* __restrict__ wbf,
                     float* __restrict__ qf, float* __restrict__ kf,
                     bf16* __restrict__ vT) {
  const int which = blockIdx.z;
  const int m0 = blockIdx.x * 128;
  const int n0 = blockIdx.y * 128;
  const bf16* W = wbf + (size_t)which * 1024 * 1024;

  __shared__ __align__(16) bf16 As[128 * 32];
  __shared__ __align__(16) bf16 Bs[128 * 32];

  const int tid = threadIdx.x;
  const int wave = tid >> 5, lane = tid & 31;
  const int wm = wave >> 1, wn = wave & 1;

  const v8f vzero = {};
  v8f acc[2][4];
#pragma unroll
  for (int i = 0; i < 2; ++i)
#pragma unroll
    for (int j = 0; j < 4; ++j) acc[i][j] = vzero;

  for (int kc = 0; kc < 1024; kc += 32) {
    __syncthreads();
#pragma unroll
    for (int r = 0; r < 2; ++r) {
      const int idx = tid + r * 256;          // 0..511 uint4 slots
      const int row = idx >> 2, c4 = idx & 3; // 8 bf16 per slot
      *(uint4*)(As + row * 32 + c4 * 8) =
          *(const uint4*)(hbf + (size_t)(m0 + row) * 1024 + kc + c4 * 8);
      *(uint4*)(Bs + row * 32 + c4 * 8) =
          *(const uint4*)(W + (size_t)(n0 + row) * 1024 + kc + c4 * 8);
    }
    __syncthreads();
    if (kc + 32 < 1024) {  // hint next A chunk -> global_prefetch_b8
      __builtin_prefetch(hbf + (size_t)(m0 + (tid >> 1)) * 1024 + kc + 32, 0, 1);
    }
    v16bf af[2], bw[4];
#pragma unroll
    for (int i = 0; i < 2; ++i)
      af[i] = load_fragA(As + (wm * 32 + i * 16) * 32, 32, lane);
#pragma unroll
    for (int j = 0; j < 4; ++j)
      bw[j] = load_fragB(Bs + (wn * 64 + j * 16) * 32, 32, lane);
#pragma unroll
    for (int i = 0; i < 2; ++i)
#pragma unroll
      for (int j = 0; j < 4; ++j)
        acc[i][j] = WMMA_BF16(false, af[i], false, bw[j], (short)0, acc[i][j],
                              false, false);
  }

  // Epilogue: C element e of v8f -> M = i*16 + (lane>>4)*8 + e, N = j*16 + (lane&15)
  const int lr = lane & 15, lh = lane >> 4;
#pragma unroll
  for (int i = 0; i < 2; ++i)
#pragma unroll
    for (int j = 0; j < 4; ++j)
#pragma unroll
      for (int e = 0; e < 8; ++e) {
        const int m = m0 + wm * 32 + i * 16 + lh * 8 + e;
        const int n = n0 + wn * 64 + j * 16 + lr;
        const int b = m >> 13, s_ = m & 8191;
        const int h = n >> 6, dd = n & 63;
        const float val = acc[i][j][e];
        if (which == 0)
          qf[((((size_t)b * 16 + h) * 8192 + s_) << 6) + dd] = val;
        else if (which == 1)
          kf[((((size_t)b * 16 + h) * 8192 + s_) << 6) + dd] = val;
        else {
          const int blk = s_ >> 8, kk = s_ & 255;
          vT[(((((size_t)b * 16 + h) * 32 + blk) * 64 + dd) << 8) + kk] =
              (bf16)val;
        }
      }
}

// ---------------------------------------------------------------- kernel 4
// Per-head LayerNorm (Dh=64) of q and k, cast to bf16. One wave per token.
__global__ __launch_bounds__(256)
void qkln_kernel(const float* __restrict__ qf, const float* __restrict__ kf,
                 bf16* __restrict__ qb, bf16* __restrict__ kb) {
  const int wave = threadIdx.x >> 5, lane = threadIdx.x & 31;
  const size_t row = (size_t)blockIdx.x * 8 + wave;  // over B*H*S = 524288
#pragma unroll
  for (int t = 0; t < 2; ++t) {
    const float* p = (t == 0 ? qf : kf) + row * 64;
    float a0 = p[lane], a1 = p[lane + 32];
    float s = a0 + a1;
    for (int off = 16; off >= 1; off >>= 1) s += __shfl_xor(s, off, 32);
    const float mean = s * (1.0f / 64.0f);
    const float d0 = a0 - mean, d1 = a1 - mean;
    float v = d0 * d0 + d1 * d1;
    for (int off = 16; off >= 1; off >>= 1) v += __shfl_xor(v, off, 32);
    const float rstd = rsqrtf(v * (1.0f / 64.0f) + 1e-5f);
    bf16* o = (t == 0 ? qb : kb) + row * 64;
    o[lane] = (bf16)(d0 * rstd);
    o[lane + 32] = (bf16)(d1 * rstd);
  }
}

// ---------------------------------------------------------------- kernel 5
// Block-local flash attention. One workgroup per (blk, h, b).
// K tile (256x64 bf16, 32KB) staged to LDS by the Tensor Data Mover (2D D#,
// data_size=2B, dims 64x256, stride 64) and synchronized with TENSORcnt;
// V read from pre-transposed vT as WMMA B-matrix; P tiles round-trip through
// per-wave LDS to convert C-layout -> A-layout.
__global__ __launch_bounds__(256)
void attn_kernel(const bf16* __restrict__ qb, const bf16* __restrict__ kb,
                 const bf16* __restrict__ vT, bf16* __restrict__ ob) {
  const int blk = blockIdx.x, h = blockIdx.y, b = blockIdx.z;
  const size_t head = (size_t)b * 16 + h;
  const bf16* qbase = qb + (head * 8192 + (size_t)blk * 256) * 64;
  const bf16* kbase = kb + (head * 8192 + (size_t)blk * 256) * 64;
  const bf16* vbase = vT + ((head * 32 + blk) * 64) * 256;  // [64][256]
  bf16* obase = ob + (head * 8192 + (size_t)blk * 256) * 64;

  __shared__ __align__(16) bf16 Ks[256 * 64];    // 32 KB
  __shared__ __align__(16) bf16 Ps[8 * 16 * 32]; // 8 KB, per-wave scratch

  const int tid = threadIdx.x, wave = tid >> 5, lane = tid & 31;

#if HAVE_TDM
  if (wave == 0) {
    // Tensor DMA descriptor (CDNA5 ISA):
    // 2D tile: 256 rows x 64 elems, 2B elements, row stride 64 elems.
    const unsigned lds_off = (unsigned)(size_t)(void*)Ks;   // LDS byte offset
    const unsigned long long ga = (unsigned long long)(size_t)kbase;
    u32x4 g0;
    g0[0] = 1u;                                       // count=1 (valid, user)
    g0[1] = lds_off;                                  // lds_addr
    g0[2] = (unsigned)(ga & 0xffffffffu);             // global_addr[31:0]
    g0[3] = (unsigned)((ga >> 32) & 0x01ffffffu)      // global_addr[56:32]
            | (2u << 30);                             // type=2 ("image")
    i32x8 g1;
    g1[0] = (int)(1u << 16);      // wg_mask=0, data_size=1 (2 bytes)
    g1[1] = (int)(64u << 16);     // tensor_dim0 = 64 (low 16 bits)
    g1[2] = (int)(256u << 16);    // tensor_dim0 hi=0, tensor_dim1 lo = 256
    g1[3] = (int)(64u << 16);     // tensor_dim1 hi=0, tile_dim0 = 64
    g1[4] = 256;                  // tile_dim1 = 256, tile_dim2 = 0
    g1[5] = 64;                   // tensor_dim0_stride = 64
    g1[6] = 0;                    // stride hi / tensor_dim1_stride lo
    g1[7] = 0;
    i32x4 g2 = {0, 0, 0, 0};      // 2D tensor: groups 2/3 unused
    i32x4 g3 = {0, 0, 0, 0};
    i32x8 g4 = {0, 0, 0, 0, 0, 0, 0, 0};  // 6-arg toolchain: extra group, unused
    __builtin_amdgcn_tensor_load_to_lds(g0, g1, g2, g3, g4, 0);
    __builtin_amdgcn_s_wait_tensorcnt(0);             // s_wait_tensorcnt 0
  }
#else
#pragma unroll
  for (int r = 0; r < 8; ++r) {
    const int idx = tid + r * 256;  // 2048 uint4 slots = 256x64 bf16
    *(uint4*)(Ks + idx * 8) = *(const uint4*)(kbase + idx * 8);
  }
#endif
  __syncthreads();

  bf16* pw = Ps + wave * (16 * 32);
  const int lr = lane & 15, lh = lane >> 4;
  const v8f vzero = {};

  for (int mt = 0; mt < 2; ++mt) {                 // 2 M-tiles of 16 per wave
    const int m0 = (wave * 2 + mt) * 16;
    const v16bf aq0 = load_fragA(qbase + m0 * 64, 64, lane);       // Dh 0..31
    const v16bf aq1 = load_fragA(qbase + m0 * 64 + 32, 64, lane);  // Dh 32..63

    v8f o[4];
    float mrow[8], lrow[8];
#pragma unroll
    for (int d2 = 0; d2 < 4; ++d2) o[d2] = vzero;
#pragma unroll
    for (int e = 0; e < 8; ++e) { mrow[e] = -1e30f; lrow[e] = 0.f; }

    for (int c = 0; c < 256; c += 32) {            // 8 key chunks of 32
      v8f sc[2];
#pragma unroll
      for (int t = 0; t < 2; ++t) {
        v8f z = vzero;
        const v16bf bk0 = load_fragB(Ks + (c + t * 16) * 64, 64, lane);
        const v16bf bk1 = load_fragB(Ks + (c + t * 16) * 64 + 32, 64, lane);
        z = WMMA_BF16(false, aq0, false, bk0, (short)0, z, false, false);
        z = WMMA_BF16(false, aq1, false, bk1, (short)0, z, false, false);
        sc[t] = z;
      }
      // Online softmax. Row r of a C tile lives across 16 lanes of a half,
      // so xor-shuffles with offsets 8..1 reduce within each half.
      float corr[8];
#pragma unroll
      for (int e = 0; e < 8; ++e) {
        const float s0 = sc[0][e] * 0.125f;        // 1/sqrt(64)
        const float s1 = sc[1][e] * 0.125f;
        float mx = fmaxf(s0, s1);
        for (int off = 8; off >= 1; off >>= 1)
          mx = fmaxf(mx, __shfl_xor(mx, off, 32));
        const float mn = fmaxf(mrow[e], mx);
        corr[e] = __expf(mrow[e] - mn);
        const float p0 = __expf(s0 - mn), p1 = __expf(s1 - mn);
        sc[0][e] = p0; sc[1][e] = p1;
        float rs = p0 + p1;
        for (int off = 8; off >= 1; off >>= 1) rs += __shfl_xor(rs, off, 32);
        lrow[e] = lrow[e] * corr[e] + rs;
        mrow[e] = mn;
      }
#pragma unroll
      for (int d2 = 0; d2 < 4; ++d2)
#pragma unroll
        for (int e = 0; e < 8; ++e) o[d2][e] *= corr[e];

      __syncthreads();
      // Store P tile [16 rows][32 kk] bf16 to this wave's LDS scratch.
#pragma unroll
      for (int t = 0; t < 2; ++t)
#pragma unroll
        for (int e = 0; e < 8; ++e)
          pw[(lh * 8 + e) * 32 + t * 16 + lr] = (bf16)sc[t][e];
      __syncthreads();
      const v16bf ap = load_fragA(pw, 32, lane);   // P as A-matrix 16x32
#pragma unroll
      for (int d2 = 0; d2 < 4; ++d2) {
        const v16bf bv = load_fragB(vbase + (d2 * 16) * 256 + c, 256, lane);
        o[d2] = WMMA_BF16(false, ap, false, bv, (short)0, o[d2], false, false);
      }
    }
    // Normalize and store O (bf16, [B,H,S,64]).
#pragma unroll
    for (int d2 = 0; d2 < 4; ++d2)
#pragma unroll
      for (int e = 0; e < 8; ++e) {
        const int m = m0 + lh * 8 + e;
        obase[(size_t)m * 64 + d2 * 16 + lr] = (bf16)(o[d2][e] / lrow[e]);
      }
  }
}

// ---------------------------------------------------------------- kernel 6
// Output projection + residual: out = x + O x Wo^T (NT GEMM, f32 epilogue).
__global__ __launch_bounds__(256)
void proj_kernel(const bf16* __restrict__ ob, const bf16* __restrict__ wo_bf,
                 const float* __restrict__ x, float* __restrict__ out) {
  const int m0 = blockIdx.x * 128;
  const int n0 = blockIdx.y * 128;
  __shared__ __align__(16) bf16 As[128 * 32];
  __shared__ __align__(16) bf16 Bs[128 * 32];
  const int tid = threadIdx.x, wave = tid >> 5, lane = tid & 31;
  const int wm = wave >> 1, wn = wave & 1;

  const v8f vzero = {};
  v8f acc[2][4];
#pragma unroll
  for (int i = 0; i < 2; ++i)
#pragma unroll
    for (int j = 0; j < 4; ++j) acc[i][j] = vzero;

  for (int kc = 0; kc < 1024; kc += 32) {
    __syncthreads();
#pragma unroll
    for (int r = 0; r < 2; ++r) {
      const int idx = tid + r * 256;
      const int row = idx >> 2, c4 = idx & 3;
      const int m = m0 + row;
      const int b = m >> 13, s_ = m & 8191;
      const int k = kc + c4 * 8;
      const int hh = k >> 6, dd = k & 63;  // K-chunk of 32 stays in one head
      *(uint4*)(As + row * 32 + c4 * 8) =
          *(const uint4*)(ob + ((((size_t)b * 16 + hh) * 8192 + s_) << 6) + dd);
      *(uint4*)(Bs + row * 32 + c4 * 8) =
          *(const uint4*)(wo_bf + (size_t)(n0 + row) * 1024 + k);
    }
    __syncthreads();
    v16bf af[2], bw[4];
#pragma unroll
    for (int i = 0; i < 2; ++i)
      af[i] = load_fragA(As + (wm * 32 + i * 16) * 32, 32, lane);
#pragma unroll
    for (int j = 0; j < 4; ++j)
      bw[j] = load_fragB(Bs + (wn * 64 + j * 16) * 32, 32, lane);
#pragma unroll
    for (int i = 0; i < 2; ++i)
#pragma unroll
      for (int j = 0; j < 4; ++j)
        acc[i][j] = WMMA_BF16(false, af[i], false, bw[j], (short)0, acc[i][j],
                              false, false);
  }

  const int lr = lane & 15, lh = lane >> 4;
#pragma unroll
  for (int i = 0; i < 2; ++i)
#pragma unroll
    for (int j = 0; j < 4; ++j)
#pragma unroll
      for (int e = 0; e < 8; ++e) {
        const int m = m0 + wm * 32 + i * 16 + lh * 8 + e;
        const int n = n0 + wn * 64 + j * 16 + lr;
        const size_t idx = (size_t)m * 1024 + n;
        out[idx] = x[idx] + acc[i][j][e];
      }
}

// ---------------------------------------------------------------- launch
extern "C" void kernel_launch(void* const* d_in, const int* in_sizes, int n_in,
                              void* d_out, int out_size, void* d_ws,
                              size_t ws_size, hipStream_t stream) {
  const float* x  = (const float*)d_in[0];
  const float* Wq = (const float*)d_in[1];
  const float* Wk = (const float*)d_in[2];
  const float* Wv = (const float*)d_in[3];
  const float* Wo = (const float*)d_in[4];
  float* out = (float*)d_out;

  // Workspace layout (bytes), ~584 MB total.
  char* ws = (char*)d_ws;
  bf16*  hbf = (bf16*)(ws);                           // 32768*1024 bf16 = 64MB
  bf16*  wbf = (bf16*)(ws + 67108864ull);             // 4*1M bf16      =  8MB
  float* qf  = (float*)(ws + 75497472ull);            // [B,H,S,64] f32 = 128MB
  float* kf  = (float*)(ws + 209715200ull);           //                = 128MB
  bf16*  vT  = (bf16*)(ws + 343932928ull);            // [B,H,32,64,256]=  64MB
  bf16*  qbb = (bf16*)(ws + 411041792ull);            //                =  64MB
  bf16*  kbb = (bf16*)(ws + 478150656ull);            //                =  64MB
  bf16*  obb = (bf16*)(ws + 545259520ull);            //                =  64MB

  ln_cast_kernel<<<32768, 256, 0, stream>>>(x, hbf);
  wcast_kernel<<<4096, 256, 0, stream>>>(Wq, Wk, Wv, Wo, wbf);
  qkv_gemm_kernel<<<dim3(256, 8, 3), 256, 0, stream>>>(hbf, wbf, qf, kf, vT);
  qkln_kernel<<<65536, 256, 0, stream>>>(qf, kf, qbb, kbb);
  attn_kernel<<<dim3(32, 16, 4), 256, 0, stream>>>(qbb, kbb, vT, obb);
  proj_kernel<<<dim3(256, 8), 256, 0, stream>>>(
      obb, wbf + (size_t)3 * 1024 * 1024, x, out);
}
